// SparseAttention_77421080478014
// MI455X (gfx1250) — matrix operand order repaired and checked by reading, not compiled, not run on hardware
//
#include <hip/hip_runtime.h>

// ---------------------------------------------------------------------------
// SparseAttention for MI455X (gfx1250, wave32, WMMA 16x16x32 f16)
// B=2, S=2048, D=1024, H=16, HD=64, SPARSITY=0.5 -> keep k=1024 per row
// ---------------------------------------------------------------------------

typedef __attribute__((ext_vector_type(16))) _Float16 v16h;
typedef __attribute__((ext_vector_type(8)))  _Float16 v8h;
typedef __attribute__((ext_vector_type(4)))  _Float16 v4h;
typedef __attribute__((ext_vector_type(8)))  float    v8f;
typedef __attribute__((ext_vector_type(4)))  int      v4i;

#define Bc    2
#define Sc    2048
#define Dc    1024
#define Hc    16
#define HDc   64
#define KKEEP 1024
#define SCALE 0.125f  // 1/sqrt(64)

// -------------------- CDNA5 async global->LDS copy (guarded) ---------------
#if defined(__AMDGCN__) && __has_builtin(__builtin_amdgcn_global_load_async_to_lds_b128)
#define HAVE_ASYNC_LDS 1
#else
#define HAVE_ASYNC_LDS 0
#endif

__device__ __forceinline__ void cp16_g2l(const _Float16* g, _Float16* l) {
#if HAVE_ASYNC_LDS
  // ROCm 7.2 signature (from diagnostics): param 1 is a generic int4*.
  __builtin_amdgcn_global_load_async_to_lds_b128(
      (v4i*)g,
      (__attribute__((address_space(3))) v4i*)l,
      0, 0);
#else
  *(v8h*)l = *(const v8h*)g;
#endif
}
__device__ __forceinline__ void cp_wait_all() {
#if HAVE_ASYNC_LDS
#if __has_builtin(__builtin_amdgcn_s_wait_asynccnt)
  __builtin_amdgcn_s_wait_asynccnt(0);
#else
  asm volatile("s_wait_asynccnt 0" ::: "memory");
#endif
#endif
}

// -------------------- WMMA helpers -----------------------------------------
__device__ __forceinline__ v8f wmma16(v16h a, v16h b, v8f c) {
  return __builtin_amdgcn_wmma_f32_16x16x32_f16(
      false, a, false, b, (short)0, c, false, false);
}

// A-frag 16x32 f16: lane (m=l&15, hi=l>>4) holds contiguous K run [hi*16..+15]
__device__ __forceinline__ v16h afrag(const _Float16* p, int hi) {
  return *(const v16h*)(p + hi * 16);
}
// B-frag 32x16 f16: lane (n=l&15, hi) holds contiguous K run [hi*16..+15]
__device__ __forceinline__ v16h bfrag(const _Float16* p, int hi) {
  return *(const v16h*)(p + hi * 16);
}

// order-preserving float<->uint key
__device__ __forceinline__ unsigned key_of(float f) {
  unsigned u = __float_as_uint(f);
  return (u & 0x80000000u) ? ~u : (u | 0x80000000u);
}
__device__ __forceinline__ float key_to_float(unsigned k) {
  unsigned u = (k & 0x80000000u) ? (k & 0x7fffffffu) : ~k;
  return __uint_as_float(u);
}

// ---------------------------------------------------------------------------
// f32 -> f16 conversion (one-time; removes per-tile converts from GEMMs)
// ---------------------------------------------------------------------------
__global__ void __launch_bounds__(256)
cvt_f32_f16(const float* __restrict__ s, _Float16* __restrict__ d, int n4) {
  const int i = blockIdx.x * 256 + threadIdx.x;
  if (i < n4) {
    const float4 v = ((const float4*)s)[i];
    v4h o;
    o[0] = (_Float16)v.x; o[1] = (_Float16)v.y;
    o[2] = (_Float16)v.z; o[3] = (_Float16)v.w;
    ((v4h*)d)[i] = o;
  }
}

// ---------------------------------------------------------------------------
// Projection GEMM: C[M,N] = A[M,K] * W[N,K]^T + bias[N];  M=4096, N=K=1024.
// A, W are f16 (pre-converted).  LDS-staged tiles (async copies on CDNA5).
// OMODE 0: f16 head-major  Qh/Kh[((b*H+h)*S+s)*HD+hd]
// OMODE 1: f16 transposed  VTh[((b*H+h)*HD+hd)*S+s]
// OMODE 2: f32 row-major   out_f[m*D+n]
// Block = 128 threads (4 waves); tile 64(M)x64(N); wave w -> rows m0+w*16.
// ---------------------------------------------------------------------------
template <int OMODE>
__global__ void __launch_bounds__(128)
proj_kernel(const _Float16* __restrict__ A, const _Float16* __restrict__ W,
            const float* __restrict__ bias, _Float16* __restrict__ out_h,
            float* __restrict__ out_f) {
  __shared__ _Float16 At[64][32];
  __shared__ _Float16 Bt[64][32];

  const int tid  = threadIdx.x;
  const int wave = tid >> 5;
  const int lane = tid & 31;
  const int hi   = lane >> 4;
  const int ln   = lane & 15;
  const int n0   = blockIdx.x * 64;
  const int mblk = blockIdx.y * 64;

  const _Float16* Ag = A + (size_t)mblk * Dc;
  const _Float16* Wg = W + (size_t)n0 * Dc;

  v8f acc[4] = {{}, {}, {}, {}};

  for (int kb = 0; kb < Dc; kb += 32) {
    __syncthreads();
#pragma unroll
    for (int rep = 0; rep < 2; ++rep) {          // 256 chunks of 8 halfs each
      const int c   = tid + rep * 128;
      const int r   = c >> 2;
      const int col = (c & 3) * 8;
      cp16_g2l(Ag + (size_t)r * Dc + kb + col, &At[r][col]);
      cp16_g2l(Wg + (size_t)r * Dc + kb + col, &Bt[r][col]);
    }
    cp_wait_all();
    __syncthreads();
    v16h a = afrag(&At[wave * 16 + ln][0], hi);
#pragma unroll
    for (int j = 0; j < 4; ++j) {
      v16h b = bfrag(&Bt[j * 16 + ln][0], hi);
      acc[j] = wmma16(a, b, acc[j]);
    }
  }

  const int m0 = mblk + wave * 16;
#pragma unroll
  for (int j = 0; j < 4; ++j) {
    const int n = n0 + j * 16 + ln;
    const float bv = bias[n];
#pragma unroll
    for (int r = 0; r < 8; ++r) {
      const int m = m0 + r + 8 * hi;
      const float v = acc[j][r] + bv;
      if (OMODE == 2) {
        out_f[(size_t)m * Dc + n] = v;
      } else {
        const int b = m >> 11, s = m & (Sc - 1);
        const int h = n >> 6,  hd = n & (HDc - 1);
        if (OMODE == 0)
          out_h[(((size_t)(b * Hc + h) * Sc + s) << 6) + hd] = (_Float16)v;
        else
          out_h[(((size_t)(b * Hc + h) * HDc + hd) << 11) + s] = (_Float16)v;
      }
    }
  }
}

// ---------------------------------------------------------------------------
// Attention: block = (16 q-rows, bh). 128 threads / 4 waves.
// LDS: 128KB f32 scores + 64KB f16 probs + 16KB radix hist (~211KB of 320KB).
// ---------------------------------------------------------------------------
struct AttnSmem {
  float    sc[16][Sc];       // 131072 B : scores, then unnormalized exp
  _Float16 pf[16][Sc];       //  65536 B : normalized probs (PV A-matrix)
  int      hist[16][256];    //  16384 B : radix-select histograms
  _Float16 qt[16][HDc];      //   2048 B : Q tile
  float    red_f[16][8];
  unsigned selp[16];
  int      selk[16];
};

__global__ void __launch_bounds__(128)
attn_kernel(const _Float16* __restrict__ Qh, const _Float16* __restrict__ Kh,
            const _Float16* __restrict__ VTh, float* __restrict__ attn_out,
            _Float16* __restrict__ AOh) {
  __shared__ AttnSmem sm;

  const int tid  = threadIdx.x;
  const int wave = tid >> 5;
  const int lane = tid & 31;
  const int hi   = lane >> 4;
  const int ln   = lane & 15;
  const int q0   = blockIdx.x * 16;
  const int bh   = blockIdx.y;                      // b*H + h
  const size_t baseQK = (size_t)bh * Sc * HDc;      // [bh][s][hd]

  // ---- stage Q tile (async on CDNA5) -----------------------------------
  {
    const int r = tid >> 3, col = (tid & 7) * 8;    // 128 chunks of 8 halfs
    cp16_g2l(Qh + baseQK + (size_t)(q0 + r) * HDc + col, &sm.qt[r][col]);
    cp_wait_all();
  }
  __syncthreads();

  const v16h qa0 = afrag(&sm.qt[ln][0], hi);
  const v16h qa1 = afrag(&sm.qt[ln][32], hi);

  // ---- scores = Q K^T * scale -> LDS -----------------------------------
  for (int nt = wave; nt < Sc / 16; nt += 4) {
    v8f acc = {};
    const _Float16* krow = Kh + baseQK + (size_t)(nt * 16 + ln) * HDc;
    acc = wmma16(qa0, bfrag(krow, hi), acc);
    acc = wmma16(qa1, bfrag(krow + 32, hi), acc);
#pragma unroll
    for (int r = 0; r < 8; ++r)
      sm.sc[r + 8 * hi][nt * 16 + ln] = acc[r] * SCALE;
  }
  __syncthreads();

  // ---- exact k-th largest per row: radix-256 select (4 passes) ---------
  const int row = tid >> 3, sub = tid & 7;          // 8 threads / row
  unsigned prefix = 0u;
  int kneed = KKEEP;
  float mx = -3.0e38f;
  for (int pass = 0; pass < 4; ++pass) {
    const int shift = 24 - 8 * pass;
    const unsigned himask = pass ? (0xFFFFFFFFu << (shift + 8)) : 0u;
    for (int u = sub; u < 256; u += 8) sm.hist[row][u] = 0;
    __syncthreads();
    for (int j = 0; j < Sc / 32; ++j) {
      const int c = j * 32 + sub * 4;
      const float4 s4 = *(const float4*)&sm.sc[row][c];
      const float sv[4] = {s4.x, s4.y, s4.z, s4.w};
#pragma unroll
      for (int e = 0; e < 4; ++e) {
        if (pass == 0) mx = fmaxf(mx, sv[e]);
        const unsigned key = key_of(sv[e]);
        if ((key & himask) == prefix)
          atomicAdd(&sm.hist[row][(key >> shift) & 255], 1);
      }
    }
    __syncthreads();
    if (sub == 0) {
      int cum = 0, d = 255;
      for (; d >= 0; --d) {
        const int cnt = sm.hist[row][d];
        if (cum + cnt >= kneed) break;
        cum += cnt;
      }
      if (d < 0) d = 0;
      sm.selp[row] = prefix | ((unsigned)d << shift);
      sm.selk[row] = kneed - cum;
    }
    __syncthreads();
    prefix = sm.selp[row];
    kneed  = sm.selk[row];
  }
  const float thresh = key_to_float(prefix);

  // ---- row max reduce --------------------------------------------------
  sm.red_f[row][sub] = mx;
  __syncthreads();
  float rmax = -3.0e38f;
#pragma unroll
  for (int u = 0; u < 8; ++u) rmax = fmaxf(rmax, sm.red_f[row][u]);
  __syncthreads();

  // ---- masked exp (stored unnormalized) + sum reduce -------------------
  float smm = 0.f;
  for (int j = 0; j < Sc / 32; ++j) {
    const int c = j * 32 + sub * 4;
    float4 s4 = *(const float4*)&sm.sc[row][c];
    s4.x = (s4.x >= thresh) ? __expf(s4.x - rmax) : 0.f;
    s4.y = (s4.y >= thresh) ? __expf(s4.y - rmax) : 0.f;
    s4.z = (s4.z >= thresh) ? __expf(s4.z - rmax) : 0.f;
    s4.w = (s4.w >= thresh) ? __expf(s4.w - rmax) : 0.f;
    *(float4*)&sm.sc[row][c] = s4;
    smm += s4.x + s4.y + s4.z + s4.w;
  }
  sm.red_f[row][sub] = smm;
  __syncthreads();
  float rsum = 0.f;
#pragma unroll
  for (int u = 0; u < 8; ++u) rsum += sm.red_f[row][u];
  const float inv = 1.0f / rsum;

  // ---- normalize: write attn (HBM, vectorized) + f16 prob tile ---------
  const size_t aoff = ((size_t)bh * Sc + (q0 + row)) * Sc;
  for (int j = 0; j < Sc / 32; ++j) {
    const int c = j * 32 + sub * 4;
    const float4 e4 = *(const float4*)&sm.sc[row][c];
    float4 p4;
    p4.x = e4.x * inv; p4.y = e4.y * inv;
    p4.z = e4.z * inv; p4.w = e4.w * inv;
    *(float4*)&attn_out[aoff + c] = p4;
    v4h ph;
    ph[0] = (_Float16)p4.x; ph[1] = (_Float16)p4.y;
    ph[2] = (_Float16)p4.z; ph[3] = (_Float16)p4.w;
    *(v4h*)&sm.pf[row][c] = ph;
  }
  __syncthreads();

  // ---- out = P @ V   (wave w -> head-dim block w*16) -------------------
  v8f oacc = {};
  const _Float16* vrow =
      VTh + (size_t)bh * HDc * Sc + (size_t)(wave * 16 + ln) * Sc;
  for (int kb = 0; kb < Sc; kb += 32) {
    v16h pa = afrag(&sm.pf[ln][kb], hi);
    v16h vb = bfrag(vrow + kb, hi);
    oacc = wmma16(pa, vb, oacc);
  }

  const int bb = bh >> 4, hh = bh & (Hc - 1);
#pragma unroll
  for (int r = 0; r < 8; ++r) {
    const int m   = r + 8 * hi;
    const int tok = bb * Sc + q0 + m;
    const int ft  = hh * HDc + wave * 16 + ln;
    AOh[(size_t)tok * Dc + ft] = (_Float16)oacc[r];
  }
}

// ---------------------------------------------------------------------------
extern "C" void kernel_launch(void* const* d_in, const int* in_sizes, int n_in,
                              void* d_out, int out_size, void* d_ws,
                              size_t ws_size, hipStream_t stream) {
  const float* query = (const float*)d_in[0];
  const float* key   = (const float*)d_in[1];
  const float* value = (const float*)d_in[2];
  const float* Wq = (const float*)d_in[3];
  const float* bq = (const float*)d_in[4];
  const float* Wk = (const float*)d_in[5];
  const float* bk = (const float*)d_in[6];
  const float* Wv = (const float*)d_in[7];
  const float* bv = (const float*)d_in[8];
  const float* Wo = (const float*)d_in[9];
  const float* bo = (const float*)d_in[10];

  float* out  = (float*)d_out;                 // (B,S,D)
  float* attn = out + (size_t)Bc * Sc * Dc;    // (B,H,S,S)

  const size_t NE = (size_t)Bc * Sc * Dc;      // 4,194,304
  const size_t WE = (size_t)Dc * Dc;           // 1,048,576
  _Float16* Xq  = (_Float16*)d_ws;
  _Float16* Xk  = Xq + NE;
  _Float16* Xv  = Xk + NE;
  _Float16* W16q = Xv + NE;
  _Float16* W16k = W16q + WE;
  _Float16* W16v = W16k + WE;
  _Float16* W16o = W16v + WE;
  _Float16* Qh  = W16o + WE;
  _Float16* Kh  = Qh + NE;
  _Float16* VTh = Kh + NE;
  _Float16* AOh = VTh + NE;                    // total 64 MB of ws

  const int n4x = (int)(NE / 4), n4w = (int)(WE / 4);
  cvt_f32_f16<<<(n4x + 255) / 256, 256, 0, stream>>>(query, Xq, n4x);
  cvt_f32_f16<<<(n4x + 255) / 256, 256, 0, stream>>>(key,   Xk, n4x);
  cvt_f32_f16<<<(n4x + 255) / 256, 256, 0, stream>>>(value, Xv, n4x);
  cvt_f32_f16<<<(n4w + 255) / 256, 256, 0, stream>>>(Wq, W16q, n4w);
  cvt_f32_f16<<<(n4w + 255) / 256, 256, 0, stream>>>(Wk, W16k, n4w);
  cvt_f32_f16<<<(n4w + 255) / 256, 256, 0, stream>>>(Wv, W16v, n4w);
  cvt_f32_f16<<<(n4w + 255) / 256, 256, 0, stream>>>(Wo, W16o, n4w);

  dim3 pgrid(Dc / 64, (Bc * Sc) / 64);         // 16 x 64
  proj_kernel<0><<<pgrid, 128, 0, stream>>>(Xq, W16q, bq, Qh, nullptr);
  proj_kernel<0><<<pgrid, 128, 0, stream>>>(Xk, W16k, bk, Kh, nullptr);
  proj_kernel<1><<<pgrid, 128, 0, stream>>>(Xv, W16v, bv, VTh, nullptr);

  dim3 agrid(Sc / 16, Bc * Hc);                // 128 x 32
  attn_kernel<<<agrid, 128, 0, stream>>>(Qh, Kh, VTh, attn, AOh);

  proj_kernel<2><<<pgrid, 128, 0, stream>>>(AOh, W16o, bo, nullptr, out);
}
